// BiLSTMLayer_74783970558046
// MI455X (gfx1250) — matrix operand order repaired and checked by reading, not compiled, not run on hardware
//
#include <hip/hip_runtime.h>

#define T_STEPS 1024
#define BATCH   64
#define DIN     128
#define HID     256
#define G4      (4 * HID)       // 1024 gate columns
#define KTOT    (DIN + HID)     // 384 fused K (x | h)
#define XH_STRIDE 392           // padded ushort row stride (2-way max LDS conflicts)

typedef __attribute__((ext_vector_type(16))) __bf16 v16bf;
typedef __attribute__((ext_vector_type(8)))  float  v8f;

__device__ __forceinline__ unsigned short f2bf(float f) {
  union { float f; unsigned u; } v; v.f = f;
  unsigned r = v.u + 0x7fffu + ((v.u >> 16) & 1u);   // round-to-nearest-even
  return (unsigned short)(r >> 16);
}
__device__ __forceinline__ float sigmoidf_(float x) { return 1.f / (1.f + __expf(-x)); }
__device__ __forceinline__ float tanhf_(float x)    { return 2.f / (1.f + __expf(-2.f * x)) - 1.f; }

// ---------------- pack kernels (run once per launch, fully parallel) --------

__global__ void pack_x_kernel(const float* __restrict__ x,
                              unsigned short* __restrict__ xbf, int n) {
  int i = blockIdx.x * blockDim.x + threadIdx.x;
  if (i < n) xbf[i] = f2bf(x[i]);
}

// wcomb[dir][j][k] = k < DIN ? Wih[dir][j][k] : Whh[dir][j][k-DIN]   (bf16)
__global__ void pack_w_kernel(const float* __restrict__ Wih_f,
                              const float* __restrict__ Whh_f,
                              const float* __restrict__ Wih_r,
                              const float* __restrict__ Whh_r,
                              unsigned short* __restrict__ wcomb) {
  int i = blockIdx.x * blockDim.x + threadIdx.x;
  if (i >= 2 * G4 * KTOT) return;
  int dir = i / (G4 * KTOT);
  int rem = i - dir * (G4 * KTOT);
  int j = rem / KTOT;
  int k = rem - j * KTOT;
  const float* Wih = dir ? Wih_r : Wih_f;
  const float* Whh = dir ? Whh_r : Whh_f;
  float v = (k < DIN) ? Wih[j * DIN + k] : Whh[j * HID + (k - DIN)];
  wcomb[i] = f2bf(v);
}

__global__ void pack_bias_kernel(const float* __restrict__ bih_f,
                                 const float* __restrict__ bhh_f,
                                 const float* __restrict__ bih_r,
                                 const float* __restrict__ bhh_r,
                                 float* __restrict__ bsum) {
  int i = blockIdx.x * blockDim.x + threadIdx.x;
  if (i >= 2 * G4) return;
  int dir = i >> 10;
  int j = i & (G4 - 1);
  bsum[i] = dir ? (bih_r[j] + bhh_r[j]) : (bih_f[j] + bhh_f[j]);
}

// ---------------- persistent bidirectional LSTM scan ------------------------
// grid = (4 batch tiles, 2 directions), 512 threads = 16 wave32.
// Wave w owns h-columns [16w, 16w+16) for all 4 gates (4 WMMA accumulators),
// so i,f,g,o land in-register with matching lane layout for the cell update.

__global__ __launch_bounds__(512, 1)
void bilstm_scan_kernel(const unsigned short* __restrict__ xbf,   // [T][B][DIN] bf16
                        const unsigned short* __restrict__ wcomb, // [2][G4][KTOT] bf16
                        const float* __restrict__ bsum,           // [2][G4]
                        const float* __restrict__ hx,             // [2][B][HID]
                        const float* __restrict__ cx,             // [2][B][HID]
                        float* __restrict__ out)                  // out|hy|cy
{
  __shared__ __align__(16) unsigned short xh[16 * XH_STRIDE];     // [16 rows][x(128)|h(256)]

  const int tid  = threadIdx.x;
  const int wave = tid >> 5;
  const int lane = tid & 31;
  const int n    = lane & 15;        // column within 16x16 tile
  const int half = lane >> 4;        // row/K half select per WMMA layout
  const int b0   = blockIdx.x * 16;  // batch tile base
  const int dir  = blockIdx.y;
  const int hcol = wave * 16 + n;    // 0..255 hidden column owned by this lane

  // Per-gate weight row bases + fused bias
  size_t wrow[4];
  float  bias4[4];
#pragma unroll
  for (int g = 0; g < 4; ++g) {
    int j = g * HID + hcol;
    wrow[g]  = ((size_t)dir * G4 + j) * KTOT;
    bias4[g] = bsum[dir * G4 + j];
  }

  // c-state and h in registers, laid out exactly like the WMMA C/D accumulator:
  // element r <-> (batch row = b0 + r + 8*half, hidden col = hcol)
  v8f   cst;
  float hreg[8];
#pragma unroll
  for (int r = 0; r < 8; ++r) {
    int b = b0 + r + 8 * half;
    cst[r]  = cx[((size_t)dir * BATCH + b) * HID + hcol];
    hreg[r] = hx[((size_t)dir * BATCH + b) * HID + hcol];
  }

  // Seed LDS xh: h part from hx, x part from the first timestep
  for (int i = tid; i < 16 * HID; i += 512) {
    int m = i >> 8, c = i & 255;
    xh[m * XH_STRIDE + DIN + c] =
        f2bf(hx[((size_t)dir * BATCH + b0 + m) * HID + c]);
  }
  {
    int t0  = (dir == 0) ? 0 : (T_STEPS - 1);
    int row = tid >> 5;
    int c4  = (tid & 31) * 4;
    const unsigned short* src = xbf + ((size_t)t0 * BATCH + b0 + row) * DIN + c4;
    *(uint2*)&xh[row * XH_STRIDE + c4] = *(const uint2*)src;
  }
  __syncthreads();

  const unsigned short* wbase = wcomb;

  for (int step = 0; step < T_STEPS; ++step) {
    const int t = (dir == 0) ? step : (T_STEPS - 1 - step);

    // Accumulators start at fused bias (broadcast down the column)
    v8f acc[4];
#pragma unroll
    for (int g = 0; g < 4; ++g) {
      v8f a;
#pragma unroll
      for (int r = 0; r < 8; ++r) a[r] = bias4[g];
      acc[g] = a;
    }

    // K loop over fused [x_t | h_{t-1}] with K=384 (12 x WMMA K=32)
    const int m = n;  // A-matrix row = lane&15 per 16-bit A layout
#pragma unroll 4
    for (int kk = 0; kk < KTOT; kk += 32) {
      // A (16x32 bf16): lane<16 holds K kk..kk+7 & kk+16..23; lane>=16 the other halves
      union { uint4 u[2]; v16bf v; } A;
      const unsigned short* ap = &xh[m * XH_STRIDE + kk + half * 8];
      A.u[0] = *(const uint4*)ap;
      A.u[1] = *(const uint4*)(ap + 16);
#pragma unroll
      for (int g = 0; g < 4; ++g) {
        // B (32x16 bf16): lane n holds column N=n, K kk+8*half..+7 & kk+16+8*half..+7
        union { uint4 u[2]; v16bf v; } Bm;
        const unsigned short* bp = wbase + wrow[g] + kk + half * 8;
        Bm.u[0] = *(const uint4*)bp;
        Bm.u[1] = *(const uint4*)(bp + 16);
        acc[g] = __builtin_amdgcn_wmma_f32_16x16x32_bf16(
            false, A.v, false, Bm.v, (short)0, acc[g], false, false);
      }
    }

    __syncthreads();  // all waves finished reading xh for this step

    // LSTM cell: everything in-register, accumulator-aligned
#pragma unroll
    for (int r = 0; r < 8; ++r) {
      float iv = sigmoidf_(acc[0][r]);
      float fv = sigmoidf_(acc[1][r]);
      float gv = tanhf_(acc[2][r]);
      float ov = sigmoidf_(acc[3][r]);
      float cv = fv * cst[r] + iv * gv;
      cst[r] = cv;
      float hv = ov * tanhf_(cv);
      hreg[r] = hv;

      int b = b0 + r + 8 * half;
      out[((size_t)t * BATCH + b) * (2 * HID) + dir * HID + hcol] = hv;
      xh[(r + 8 * half) * XH_STRIDE + DIN + hcol] = f2bf(hv);
    }

    // Stage next timestep's x tile into LDS
    if (step + 1 < T_STEPS) {
      int tn  = (dir == 0) ? (step + 1) : (T_STEPS - 2 - step);
      int row = tid >> 5;
      int c4  = (tid & 31) * 4;
      const unsigned short* src = xbf + ((size_t)tn * BATCH + b0 + row) * DIN + c4;
      *(uint2*)&xh[row * XH_STRIDE + c4] = *(const uint2*)src;
    }
    __syncthreads();
  }

  // Final states: hy then cy appended after out
  float* hy = out + (size_t)T_STEPS * BATCH * 2 * HID;
  float* cy = hy + (size_t)2 * BATCH * HID;
#pragma unroll
  for (int r = 0; r < 8; ++r) {
    int b = b0 + r + 8 * half;
    hy[((size_t)dir * BATCH + b) * HID + hcol] = hreg[r];
    cy[((size_t)dir * BATCH + b) * HID + hcol] = cst[r];
  }
}

// ---------------- host launch ----------------------------------------------

extern "C" void kernel_launch(void* const* d_in, const int* in_sizes, int n_in,
                              void* d_out, int out_size, void* d_ws, size_t ws_size,
                              hipStream_t stream) {
  const float* x     = (const float*)d_in[0];
  const float* hx    = (const float*)d_in[1];
  const float* cx    = (const float*)d_in[2];
  const float* Wih_f = (const float*)d_in[3];
  const float* Whh_f = (const float*)d_in[4];
  const float* bih_f = (const float*)d_in[5];
  const float* bhh_f = (const float*)d_in[6];
  const float* Wih_r = (const float*)d_in[7];
  const float* Whh_r = (const float*)d_in[8];
  const float* bih_r = (const float*)d_in[9];
  const float* bhh_r = (const float*)d_in[10];

  // Workspace layout (bf16 staging): xbf | wcomb | bsum  (~18.4 MB total)
  unsigned short* xbf   = (unsigned short*)d_ws;
  unsigned short* wcomb = xbf + (size_t)T_STEPS * BATCH * DIN;
  float*          bsum  = (float*)(wcomb + (size_t)2 * G4 * KTOT);

  {
    int n = T_STEPS * BATCH * DIN;
    pack_x_kernel<<<(n + 255) / 256, 256, 0, stream>>>(x, xbf, n);
  }
  {
    int n = 2 * G4 * KTOT;
    pack_w_kernel<<<(n + 255) / 256, 256, 0, stream>>>(Wih_f, Whh_f, Wih_r, Whh_r, wcomb);
  }
  pack_bias_kernel<<<(2 * G4 + 255) / 256, 256, 0, stream>>>(bih_f, bhh_f, bih_r, bhh_r, bsum);

  dim3 grid(BATCH / 16, 2);  // 4 batch tiles x 2 directions
  bilstm_scan_kernel<<<grid, 512, 0, stream>>>(xbf, wcomb, bsum, hx, cx, (float*)d_out);

  (void)in_sizes; (void)n_in; (void)out_size; (void)ws_size;
}
